// xLSTMModule_32796370272631
// MI455X (gfx1250) — matrix operand order repaired and checked by reading, not compiled, not run on hardware
//
#include <hip/hip_runtime.h>

// ---------------------------------------------------------------------------
// Types for CDNA5 WMMA (wave32, 16x16x32 bf16 -> f32)
// ---------------------------------------------------------------------------
typedef __bf16 bf16;
typedef __bf16 bf16x16v __attribute__((ext_vector_type(16)));
typedef __bf16 bf16x4v  __attribute__((ext_vector_type(4)));
typedef float  f32x8v   __attribute__((ext_vector_type(8)));

__device__ inline f32x8v wmma_bf16(bf16x16v a, bf16x16v b, f32x8v c) {
  // 8 args: (neg_a, A, neg_b, B, c_mod, C, reuse_a, reuse_b)
  return __builtin_amdgcn_wmma_f32_16x16x32_bf16(false, a, false, b, (short)0, c,
                                                 false, false);
}

__device__ inline float sigmoidf_(float x) { return 1.f / (1.f + __expf(-x)); }

// ---------------------------------------------------------------------------
// Generic GEMM: C[M,N] = A[M,K] * W[N,K]^T   (bf16 WMMA, f32 accumulate)
// M, N multiples of 64; K multiple of 32; all pointers 16B aligned.
// ACC != 0 -> C += result (fused residual), selected at compile time.
// Fragment layouts per CDNA5 ISA 7.12.2:
//   A 16x32 bf16: lane<16 row=lane, elems K {0..7,16..23}; lane>=16 {8..15,24..31}
//   B 32x16 bf16: lane holds column n=lane&15, elems K same as A (LDS kept k-major)
//   C/D f32 16x16: lane: n=lane&15, rows (lane>>4)*8 + r
// ---------------------------------------------------------------------------
template <int ACC>
__global__ __launch_bounds__(128)
void gemm_bf16_kernel(const float* __restrict__ A, int lda,
                      const float* __restrict__ Bw, int ldb,
                      float* __restrict__ C, int ldc, int Kd) {
  __shared__ bf16 lA[64][32];    // [m][k]
  __shared__ bf16 lBt[32][64];   // [k][n]  (transposed -> contiguous B fragments)
  const int bm = blockIdx.y * 64;
  const int bn = blockIdx.x * 64;
  const int tid = threadIdx.x;
  const int wave = tid >> 5, lane = tid & 31;
  const int wm = (wave & 1) * 32, wn = (wave >> 1) * 32;
  const int half = lane >> 4, hr = lane & 15;
  const int ka = half * 8, kb2 = 16 + half * 8;
  const f32x8v zero = {0.f, 0.f, 0.f, 0.f, 0.f, 0.f, 0.f, 0.f};
  f32x8v accv[2][2];
#pragma unroll
  for (int mi = 0; mi < 2; mi++)
#pragma unroll
    for (int ni = 0; ni < 2; ni++) accv[mi][ni] = zero;

  for (int k0 = 0; k0 < Kd; k0 += 32) {
    // Stage A tile (row-major, vectorized: float4 load -> bf16x4 LDS store)
    for (int i = tid; i < 512; i += 128) {
      int r = i >> 3, c4 = (i & 7) * 4;
      float4 v = *(const float4*)(A + (long)(bm + r) * lda + k0 + c4);
      bf16x4v p;
      p[0] = (bf16)v.x; p[1] = (bf16)v.y; p[2] = (bf16)v.z; p[3] = (bf16)v.w;
      *(bf16x4v*)&lA[r][c4] = p;
    }
    // Stage B tile transposed: k-major so fragments are contiguous
    for (int i = tid; i < 512; i += 128) {
      int r = i >> 3, c4 = (i & 7) * 4;
      float4 v = *(const float4*)(Bw + (long)(bn + r) * ldb + k0 + c4);
      lBt[c4 + 0][r] = (bf16)v.x;
      lBt[c4 + 1][r] = (bf16)v.y;
      lBt[c4 + 2][r] = (bf16)v.z;
      lBt[c4 + 3][r] = (bf16)v.w;
    }
    __syncthreads();
    // Software prefetch of next K-step tiles (global_prefetch_b8, no counter):
    // overlaps next tile's L2 fill with this tile's WMMAs.
    if (k0 + 32 < Kd) {
      if (tid < 64) __builtin_prefetch(A + (long)(bm + tid) * lda + k0 + 32, 0, 0);
      else          __builtin_prefetch(Bw + (long)(bn + tid - 64) * ldb + k0 + 32, 0, 0);
    }
    bf16x16v af[2], bfv[2];
#pragma unroll
    for (int i = 0; i < 2; i++) {
      int row = wm + i * 16 + hr;
#pragma unroll
      for (int j = 0; j < 8; j++) {
        af[i][j] = lA[row][ka + j];
        af[i][8 + j] = lA[row][kb2 + j];
      }
    }
#pragma unroll
    for (int i = 0; i < 2; i++) {
      int col0 = wn + i * 16;
#pragma unroll
      for (int e = 0; e < 8; e++) { bfv[i][e] = lBt[ka + e][col0 + hr]; bfv[i][8 + e] = lBt[kb2 + e][col0 + hr]; }
    }
#pragma unroll
    for (int mi = 0; mi < 2; mi++)
#pragma unroll
      for (int ni = 0; ni < 2; ni++)
        accv[mi][ni] = wmma_bf16(af[mi], bfv[ni], accv[mi][ni]);
    __syncthreads();
  }
  const int n = lane & 15, mb = (lane >> 4) * 8;
#pragma unroll
  for (int mi = 0; mi < 2; mi++)
#pragma unroll
    for (int ni = 0; ni < 2; ni++)
#pragma unroll
      for (int r = 0; r < 8; r++) {
        long off = (long)(bm + wm + mi * 16 + mb + r) * ldc + bn + wn + ni * 16 + n;
        if (ACC) C[off] += accv[mi][ni][r];
        else     C[off]  = accv[mi][ni][r];
      }
}

// ---------------------------------------------------------------------------
// LayerNorm-style _norm: one wave per row
// ---------------------------------------------------------------------------
__global__ void norm_kernel(const float* __restrict__ x, const float* __restrict__ w,
                            float* __restrict__ out, int Dc) {
  const int row = blockIdx.x, lane = threadIdx.x;
  const float* xr = x + (long)row * Dc;
  float s = 0.f, s2 = 0.f;
  for (int j = lane; j < Dc; j += 32) { float v = xr[j]; s += v; s2 += v * v; }
#pragma unroll
  for (int m = 16; m >= 1; m >>= 1) { s += __shfl_xor(s, m, 32); s2 += __shfl_xor(s2, m, 32); }
  float mean = s / Dc;
  float var = s2 / Dc - mean * mean;
  float rstd = rsqrtf(var + 1e-5f);
  float* orow = out + (long)row * Dc;
  for (int j = lane; j < Dc; j += 32) orow[j] = (xr[j] - mean) * rstd * w[j];
}

// ---------------------------------------------------------------------------
// Causal depthwise conv (K=4) + swish
// ---------------------------------------------------------------------------
__global__ void conv_swish_kernel(const float* __restrict__ x, int ldx,
                                  const float* __restrict__ w, const float* __restrict__ bias,
                                  float* __restrict__ out, int ldo,
                                  int S, int C, long total) {
  long idx = (long)blockIdx.x * 256 + threadIdx.x;
  if (idx >= total) return;
  int c = (int)(idx % C);
  long bs = idx / C;
  int s = (int)(bs % S);
  long b = bs / S;
  float a = bias[c];
#pragma unroll
  for (int k = 0; k < 4; k++) {
    int sk = s + k - 3;
    if (sk >= 0) a += x[((long)b * S + sk) * ldx + c] * w[c * 4 + k];
  }
  out[((long)b * S + s) * ldo + c] = a * sigmoidf_(a);
}

// ---------------------------------------------------------------------------
// Block-diagonal 4x4 projections (mLSTM q/k/v)
// ---------------------------------------------------------------------------
__global__ void bd4_kernel(const float* __restrict__ x, int ldx,
                           const float* __restrict__ w,
                           float* __restrict__ out, int ldo, int nb, long total) {
  long idx = (long)blockIdx.x * 256 + threadIdx.x;
  if (idx >= total) return;
  int nblk = (int)(idx % nb);
  long row = idx / nb;
  const float* xr = x + row * ldx + nblk * 4;
  float x0 = xr[0], x1 = xr[1], x2 = xr[2], x3 = xr[3];
  const float* wb = w + nblk * 16;
  float* o = out + row * ldo + nblk * 4;
#pragma unroll
  for (int k = 0; k < 4; k++)
    o[k] = x0 * wb[k * 4 + 0] + x1 * wb[k * 4 + 1] + x2 * wb[k * 4 + 2] + x3 * wb[k * 4 + 3];
}

// ---------------------------------------------------------------------------
// ig/fg gate projections: one wave per (b,s); ig/fg stored [B,NH,S]
// ---------------------------------------------------------------------------
__global__ void igfg_kernel(const float* __restrict__ Q, const float* __restrict__ K,
                            const float* __restrict__ V,
                            const float* __restrict__ igw, const float* __restrict__ igb,
                            const float* __restrict__ fgw, const float* __restrict__ fgb,
                            float* __restrict__ ig, float* __restrict__ fg, int S) {
  const int row = blockIdx.x, lane = threadIdx.x;
  const float* q = Q + (long)row * 1024;
  const float* k = K + (long)row * 1024;
  const float* v = V + (long)row * 1024;
  float pig[4] = {0.f, 0.f, 0.f, 0.f}, pfg[4] = {0.f, 0.f, 0.f, 0.f};
  for (int j = lane; j < 1024; j += 32) {
    float qv = q[j], kv = k[j], vv = v[j];
#pragma unroll
    for (int h = 0; h < 4; h++) {
      const float* wi = igw + h * 3072;
      const float* wf = fgw + h * 3072;
      pig[h] += qv * wi[j] + kv * wi[1024 + j] + vv * wi[2048 + j];
      pfg[h] += qv * wf[j] + kv * wf[1024 + j] + vv * wf[2048 + j];
    }
  }
#pragma unroll
  for (int h = 0; h < 4; h++)
#pragma unroll
    for (int m = 16; m >= 1; m >>= 1) {
      pig[h] += __shfl_xor(pig[h], m, 32);
      pfg[h] += __shfl_xor(pfg[h], m, 32);
    }
  if (lane == 0) {
    int b = row / S, s = row % S;
#pragma unroll
    for (int h = 0; h < 4; h++) {
      ig[((long)b * 4 + h) * S + s] = pig[h] + igb[h];
      fg[((long)b * 4 + h) * S + s] = pfg[h] + fgb[h];
    }
  }
}

// ---------------------------------------------------------------------------
// Decay scan per (b,h): tb[t]=ig[t]-lfc[t+1]; rmax=running max; nfl=exp(-maxD)
// ---------------------------------------------------------------------------
__global__ void mlstm_decay_kernel(const float* __restrict__ ig, const float* __restrict__ fg,
                                   float* __restrict__ tb, float* __restrict__ rmx,
                                   float* __restrict__ nfl, int BH, int S) {
  int t = blockIdx.x * blockDim.x + threadIdx.x;
  if (t >= BH) return;
  const float* igr = ig + (long)t * S;
  const float* fgr = fg + (long)t * S;
  float lf = 0.f, rm = -1e30f;
  for (int s = 0; s < S; s++) {
    float x = fgr[s];
    lf += fminf(x, 0.f) - log1pf(__expf(-fabsf(x)));   // log_sigmoid, cumulated
    float tv = igr[s] - lf;
    tb[(long)t * S + s] = tv;
    rm = fmaxf(rm, tv);
    rmx[(long)t * S + s] = rm;
    nfl[(long)t * S + s] = __expf(-lf - rm);           // exp(-maxD[s])
  }
}

// ---------------------------------------------------------------------------
// mLSTM attention: one 64-thread (2-wave) block per (b, h, 16-row tile).
// dh = 256. QK^T and P*V via bf16 WMMA; decay/mask/normalizer in VALU.
// ---------------------------------------------------------------------------
__global__ __launch_bounds__(64)
void mlstm_attn_kernel(const float* __restrict__ Q, const float* __restrict__ K,
                       const float* __restrict__ V,
                       const float* __restrict__ tb, const float* __restrict__ rmx,
                       const float* __restrict__ nfl,
                       float* __restrict__ H, int S) {
  const int s0 = blockIdx.x * 16, h = blockIdx.y, b = blockIdx.z;
  const int tid = threadIdx.x, wave = tid >> 5, lane = tid & 31;
  const int half = lane >> 4, hr = lane & 15;
  __shared__ bf16 lQ[16][256];    // [m][d]
  __shared__ bf16 lKt[256][32];   // [d][t]  (transposed -> contiguous B fragments)
  __shared__ bf16 lV[32][256];    // [t][d]
  __shared__ bf16 lC[16][32];     // weighted scores as next A tile
  __shared__ float lRS[2][16];
  const long rowbase = ((long)b * S) * 1024 + h * 256;
  const float* qp = Q + rowbase;
  const float* kp = K + rowbase;
  const float* vp = V + rowbase;
  const float* tbr = tb + ((long)b * 4 + h) * S;
  const float* rmr = rmx + ((long)b * 4 + h) * S;
  const float* nfr = nfl + ((long)b * 4 + h) * S;

  // Stage Q (scaled by dh^-0.5 = 1/16), vectorized
  for (int i = tid; i < 1024; i += 64) {
    int r = i >> 6, c4 = (i & 63) * 4;
    float4 v = *(const float4*)(qp + (long)(s0 + r) * 1024 + c4);
    bf16x4v p;
    p[0] = (bf16)(v.x * 0.0625f); p[1] = (bf16)(v.y * 0.0625f);
    p[2] = (bf16)(v.z * 0.0625f); p[3] = (bf16)(v.w * 0.0625f);
    *(bf16x4v*)&lQ[r][c4] = p;
  }
  float rmv[8], rs[8];
#pragma unroll
  for (int r = 0; r < 8; r++) { rmv[r] = rmr[s0 + r + 8 * half]; rs[r] = 0.f; }
  const f32x8v zero = {0.f, 0.f, 0.f, 0.f, 0.f, 0.f, 0.f, 0.f};
  f32x8v oacc[8];
#pragma unroll
  for (int i = 0; i < 8; i++) oacc[i] = zero;

  const int ka = half * 8, kb2 = 16 + half * 8;
  const int nchunks = (s0 + 47) >> 5;
  for (int ch = 0; ch < nchunks; ch++) {
    const int t0 = ch * 32;
    __syncthreads();  // protect LDS reuse from previous chunk's consumers
    for (int i = tid; i < 2048; i += 64) {
      int r = i >> 6, c4 = (i & 63) * 4;  // r = t index, c4 = d
      float4 v = *(const float4*)(kp + (long)(t0 + r) * 1024 + c4);
      lKt[c4 + 0][r] = (bf16)v.x;
      lKt[c4 + 1][r] = (bf16)v.y;
      lKt[c4 + 2][r] = (bf16)v.z;
      lKt[c4 + 3][r] = (bf16)v.w;
    }
    for (int i = tid; i < 2048; i += 64) {
      int r = i >> 6, c4 = (i & 63) * 4;
      float4 v = *(const float4*)(vp + (long)(t0 + r) * 1024 + c4);
      bf16x4v p;
      p[0] = (bf16)v.x; p[1] = (bf16)v.y; p[2] = (bf16)v.z; p[3] = (bf16)v.w;
      *(bf16x4v*)&lV[r][c4] = p;
    }
    __syncthreads();
    // Prefetch next chunk's K/V rows (1KB of dh per row, 128B-line granularity)
    // while this chunk's WMMAs execute. Fire-and-forget: no counter impact.
    if (ch + 1 < nchunks) {
      const long nxt = (long)(t0 + 32) * 1024;
      for (int i = tid; i < 256; i += 64) {
        int r = i >> 3, seg = i & 7;  // 32 rows x 8 x 128B
        __builtin_prefetch(kp + nxt + (long)r * 1024 + seg * 32, 0, 0);
        __builtin_prefetch(vp + nxt + (long)r * 1024 + seg * 32, 0, 0);
      }
    }
    // QK^T: wave w computes t-tile w (16 columns)
    f32x8v sacc = zero;
#pragma unroll
    for (int kk = 0; kk < 256; kk += 32) {
      bf16x16v af, bfv;
#pragma unroll
      for (int j = 0; j < 8; j++) { af[j] = lQ[hr][kk + ka + j]; af[8 + j] = lQ[hr][kk + kb2 + j]; }
#pragma unroll
      for (int e = 0; e < 8; e++) { bfv[e] = lKt[kk + ka + e][wave * 16 + hr]; bfv[8 + e] = lKt[kk + kb2 + e][wave * 16 + hr]; }
      sacc = wmma_bf16(af, bfv, sacc);
    }
    // decay + causal mask; accumulate rowsum; stash weighted scores as bf16 A-tile
    const int tcol = t0 + wave * 16 + hr;
    const float tv = tbr[tcol];
#pragma unroll
    for (int r = 0; r < 8; r++) {
      int m = r + 8 * half;
      float dm = (tcol <= s0 + m) ? __expf(tv - rmv[r]) : 0.f;
      float cv = sacc[r] * dm;
      rs[r] += cv;
      lC[m][wave * 16 + hr] = (bf16)cv;
    }
    __syncthreads();
    // P*V: wave w handles v-dims [w*128, w*128+128)
    bf16x16v af2;
#pragma unroll
    for (int j = 0; j < 8; j++) { af2[j] = lC[hr][ka + j]; af2[8 + j] = lC[hr][kb2 + j]; }
#pragma unroll
    for (int nt = 0; nt < 8; nt++) {
      int d0 = wave * 128 + nt * 16;
      bf16x16v bfv;
#pragma unroll
      for (int e = 0; e < 8; e++) { bfv[e] = lV[ka + e][d0 + hr]; bfv[8 + e] = lV[kb2 + e][d0 + hr]; }
      oacc[nt] = wmma_bf16(af2, bfv, oacc[nt]);
    }
  }
  // row sums: reduce across 16 lanes, then across the two waves via LDS
#pragma unroll
  for (int r = 0; r < 8; r++) {
    float v = rs[r];
#pragma unroll
    for (int m2 = 8; m2 >= 1; m2 >>= 1) v += __shfl_xor(v, m2, 16);
    rs[r] = v;
  }
  __syncthreads();
  if (hr == 0) {
#pragma unroll
    for (int r = 0; r < 8; r++) lRS[wave][r + 8 * half] = rs[r];
  }
  __syncthreads();
  float inv[8];
#pragma unroll
  for (int r = 0; r < 8; r++) {
    int m = r + 8 * half;
    float tot = lRS[0][m] + lRS[1][m];
    inv[r] = 1.f / (fmaxf(fabsf(tot), nfr[s0 + m]) + 1e-6f);
  }
  float* hbp = H + rowbase;
#pragma unroll
  for (int nt = 0; nt < 8; nt++) {
    int d = wave * 128 + nt * 16 + hr;
#pragma unroll
    for (int r = 0; r < 8; r++)
      hbp[(long)(s0 + r + 8 * half) * 1024 + d] = oacc[nt][r] * inv[r];
  }
}

// ---------------------------------------------------------------------------
// mLSTM epilogue: per-(b,s,head) norm over dh=256, then (hn+skip*xconv)*swish(z)
// ---------------------------------------------------------------------------
__global__ void mlstm_combine_kernel(float* __restrict__ Hb, const float* __restrict__ xz,
                                     const float* __restrict__ xconv,
                                     const float* __restrict__ onw,
                                     const float* __restrict__ skp) {
  const int row = blockIdx.x, lane = threadIdx.x;
  long bs = row >> 2;
  int h = row & 3;
  float* hr = Hb + bs * 1024 + h * 256;
  float s = 0.f, s2 = 0.f;
  for (int j = lane; j < 256; j += 32) { float v = hr[j]; s += v; s2 += v * v; }
#pragma unroll
  for (int m = 16; m >= 1; m >>= 1) { s += __shfl_xor(s, m, 32); s2 += __shfl_xor(s2, m, 32); }
  float mean = s * (1.f / 256.f);
  float var = s2 * (1.f / 256.f) - mean * mean;
  float rstd = rsqrtf(var + 1e-5f);
  for (int j = lane; j < 256; j += 32) {
    int c = h * 256 + j;
    float hn = (hr[j] - mean) * rstd * onw[c];
    float z = xz[bs * 2048 + 1024 + c];
    hr[j] = (hn + skp[c] * xconv[bs * 1024 + c]) * (z * sigmoidf_(z));
  }
}

// ---------------------------------------------------------------------------
// sLSTM recurrence: one 128-thread workgroup per (b,h); thread = state dim
// ---------------------------------------------------------------------------
__global__ __launch_bounds__(128)
void slstm_scan_kernel(const float* __restrict__ gx, const float* __restrict__ R,
                       const float* __restrict__ cellb, float* __restrict__ hs, int S) {
  const int b = blockIdx.x >> 2, h = blockIdx.x & 3;
  const int k = threadIdx.x;  // 0..127
  __shared__ float sh[128];
  const float bi = cellb[0 * 512 + h * 128 + k];
  const float bfv = cellb[1 * 512 + h * 128 + k];
  const float bz = cellb[2 * 512 + h * 128 + k];
  const float bo = cellb[3 * 512 + h * 128 + k];
  const float* Rh = R + (long)h * 128 * 4 * 128;
  float hv = 0.f, c = 0.f, n = 0.f, m = 0.f;
  for (int s = 0; s < S; s++) {
    sh[k] = hv;
    __syncthreads();
    const float* g0 = gx + ((long)(b * S + s)) * 2048 + h * 128 + k;
    float ir = g0[0] + bi, fr = g0[512] + bfv, zr = g0[1024] + bz, og = g0[1536] + bo;
    for (int d = 0; d < 128; d++) {
      float hd = sh[d];
      const float* rp = Rh + (long)d * 512 + k;
      ir += hd * rp[0];
      fr += hd * rp[128];
      zr += hd * rp[256];
      og += hd * rp[384];
    }
    float logf = fminf(fr, 0.f) - log1pf(__expf(-fabsf(fr))) + m;
    float mn = fmaxf(logf, ir);
    float iv = __expf(ir - mn), fv = __expf(logf - mn);
    c = fv * c + iv * tanhf(zr);
    n = fv * n + iv;
    m = mn;
    hv = sigmoidf_(og) * c / n;
    hs[((long)(b * S + s)) * 512 + h * 128 + k] = hv;
    __syncthreads();
  }
}

// ---------------------------------------------------------------------------
// sLSTM epilogue: per-(b,s,head) norm over dh=128 with gn_w, add into residual
// ---------------------------------------------------------------------------
__global__ void slstm_norm_add_kernel(const float* __restrict__ hs,
                                      const float* __restrict__ w,
                                      float* __restrict__ X) {
  const int row = blockIdx.x, lane = threadIdx.x;
  long bs = row >> 2;
  int h = row & 3;
  const float* hr = hs + bs * 512 + h * 128;
  float s = 0.f, s2 = 0.f;
  for (int j = lane; j < 128; j += 32) { float v = hr[j]; s += v; s2 += v * v; }
#pragma unroll
  for (int m = 16; m >= 1; m >>= 1) { s += __shfl_xor(s, m, 32); s2 += __shfl_xor(s2, m, 32); }
  float mean = s * (1.f / 128.f);
  float var = s2 * (1.f / 128.f) - mean * mean;
  float rstd = rsqrtf(var + 1e-5f);
  for (int j = lane; j < 128; j += 32)
    X[bs * 512 + h * 128 + j] += (hr[j] - mean) * rstd * w[h * 128 + j];
}

// ---------------------------------------------------------------------------
// FFN gate: relu(g) * u  (g,u split of up-projection output)
// ---------------------------------------------------------------------------
__global__ void ffn_gate_kernel(const float* __restrict__ up, float* __restrict__ out,
                                long total) {
  long idx = (long)blockIdx.x * 256 + threadIdx.x;
  if (idx >= total) return;
  long bs = idx >> 10;
  int j = (int)(idx & 1023);
  float g = up[bs * 2048 + j];
  float u = up[bs * 2048 + 1024 + j];
  out[idx] = fmaxf(g, 0.f) * u;
}

// ---------------------------------------------------------------------------
// Host orchestration
// ---------------------------------------------------------------------------
static inline void gemm(hipStream_t st, const float* A, int lda, const float* Bw, int ldb,
                        float* C, int ldc, int M, int N, int Kd, int acc) {
  dim3 g(N / 64, M / 64);
  if (acc)
    gemm_bf16_kernel<1><<<g, 128, 0, st>>>(A, lda, Bw, ldb, C, ldc, Kd);
  else
    gemm_bf16_kernel<0><<<g, 128, 0, st>>>(A, lda, Bw, ldb, C, ldc, Kd);
}

struct MParams {
  const float *ln1, *proj_up, *conv_w, *conv_b, *q_w, *k_w, *v_w;
  const float *ig_w, *ig_b, *fg_w, *fg_b, *onorm_w, *skp, *proj_down;
};
struct SParams {
  const float *ln1, *conv_w, *conv_b, *i_w, *f_w, *z_w, *o_w;
  const float *R, *cell_b, *gn_w, *ln2, *up, *down;
};

extern "C" void kernel_launch(void* const* d_in, const int* in_sizes, int n_in,
                              void* d_out, int out_size, void* d_ws, size_t ws_size,
                              hipStream_t stream) {
  (void)n_in; (void)out_size; (void)ws_size;
  const float* const* P = (const float* const*)d_in;

  // Disambiguate flatten order: insertion order puts post_norm (512) at index 1;
  // sorted-key order puts block0.conv_b (1024) there.
  const bool insertion = (in_sizes[1] == 512);

  MParams mp[2];
  SParams sp[2];
  const float* post_norm;
  if (insertion) {
    post_norm = P[1];
    const int mbase[2] = {2, 29}, sbase[2] = {16, 43};
    for (int i = 0; i < 2; i++) {
      int o = mbase[i];
      mp[i] = {P[o + 0], P[o + 1], P[o + 2], P[o + 3], P[o + 4], P[o + 5], P[o + 6],
               P[o + 7], P[o + 8], P[o + 9], P[o + 10], P[o + 11], P[o + 12], P[o + 13]};
      int q = sbase[i];
      sp[i] = {P[q + 0], P[q + 1], P[q + 2], P[q + 3], P[q + 4], P[q + 5], P[q + 6],
               P[q + 7], P[q + 8], P[q + 9], P[q + 10], P[q + 11], P[q + 12]};
    }
  } else {
    post_norm = P[55];
    const int mbase[2] = {1, 28}, sbase[2] = {15, 42};
    for (int i = 0; i < 2; i++) {
      int o = mbase[i];
      // sorted: conv_b conv_w fg_b fg_w ig_b ig_w k_w ln1 onorm_w proj_down proj_up q_w skip v_w
      mp[i].conv_b = P[o + 0]; mp[i].conv_w = P[o + 1]; mp[i].fg_b = P[o + 2];
      mp[i].fg_w = P[o + 3];   mp[i].ig_b = P[o + 4];   mp[i].ig_w = P[o + 5];
      mp[i].k_w = P[o + 6];    mp[i].ln1 = P[o + 7];    mp[i].onorm_w = P[o + 8];
      mp[i].proj_down = P[o + 9]; mp[i].proj_up = P[o + 10]; mp[i].q_w = P[o + 11];
      mp[i].skp = P[o + 12];   mp[i].v_w = P[o + 13];
      int q = sbase[i];
      // sorted: R cell_b conv_b conv_w down f_w gn_w i_w ln1 ln2 o_w up z_w
      sp[i].R = P[q + 0];     sp[i].cell_b = P[q + 1]; sp[i].conv_b = P[q + 2];
      sp[i].conv_w = P[q + 3]; sp[i].down = P[q + 4];  sp[i].f_w = P[q + 5];
      sp[i].gn_w = P[q + 6];  sp[i].i_w = P[q + 7];    sp[i].ln1 = P[q + 8];
      sp[i].ln2 = P[q + 9];   sp[i].o_w = P[q + 10];   sp[i].up = P[q + 11];
      sp[i].z_w = P[q + 12];
    }
  }

  // Workspace layout (floats). NR = B*S = 4096 rows. Total ~135 MB.
  float* W = (float*)d_ws;
  const long NR = 8L * 512;
  float* X  = W;               // residual [NR,512]
  float* T0 = X + NR * 512;    // normed   [NR,512]
  float* T1 = T0 + NR * 512;   // xz / gx / ffn-up [NR,2048]
  float* T2 = T1 + NR * 2048;  // xconv / gated    [NR,1024]
  float* Qb = T2 + NR * 1024;  // [NR,1024]
  float* Kb = Qb + NR * 1024;
  float* Vb = Kb + NR * 1024;
  float* Hb = Vb + NR * 1024;  // attention / hidden out [NR,1024]
  float* IG = Hb + NR * 1024;  // [B,NH,S]
  float* FG = IG + 32 * 512;
  float* TBb = FG + 32 * 512;
  float* RM = TBb + 32 * 512;
  float* NF = RM + 32 * 512;

  hipMemcpyAsync(X, (const void*)P[0], NR * 512 * sizeof(float),
                 hipMemcpyDeviceToDevice, stream);

  int mi = 0, si = 0;
  for (int bi = 0; bi < 4; bi++) {
    if (bi == 0 || bi == 2) {
      const MParams& p = mp[mi++];
      norm_kernel<<<(unsigned)NR, 32, 0, stream>>>(X, p.ln1, T0, 512);
      gemm(stream, T0, 512, p.proj_up, 512, T1, 2048, (int)NR, 2048, 512, 0);
      {
        long tot = NR * 1024;
        conv_swish_kernel<<<(unsigned)((tot + 255) / 256), 256, 0, stream>>>(
            T1, 2048, p.conv_w, p.conv_b, T2, 1024, 512, 1024, tot);
      }
      {
        long tot = NR * 256;
        unsigned gb = (unsigned)((tot + 255) / 256);
        bd4_kernel<<<gb, 256, 0, stream>>>(T2, 1024, p.q_w, Qb, 1024, 256, tot);
        bd4_kernel<<<gb, 256, 0, stream>>>(T2, 1024, p.k_w, Kb, 1024, 256, tot);
        bd4_kernel<<<gb, 256, 0, stream>>>(T1, 2048, p.v_w, Vb, 1024, 256, tot);
      }
      igfg_kernel<<<(unsigned)NR, 32, 0, stream>>>(Qb, Kb, Vb, p.ig_w, p.ig_b,
                                                   p.fg_w, p.fg_b, IG, FG, 512);
      mlstm_decay_kernel<<<1, 32, 0, stream>>>(IG, FG, TBb, RM, NF, 32, 512);
      {
        dim3 g(32, 4, 8);
        mlstm_attn_kernel<<<g, 64, 0, stream>>>(Qb, Kb, Vb, TBb, RM, NF, Hb, 512);
      }
      mlstm_combine_kernel<<<(unsigned)(NR * 4), 32, 0, stream>>>(Hb, T1, T2,
                                                                  p.onorm_w, p.skp);
      gemm(stream, Hb, 1024, p.proj_down, 1024, X, 512, (int)NR, 512, 1024, 1);
    } else {
      const SParams& p = sp[si++];
      norm_kernel<<<(unsigned)NR, 32, 0, stream>>>(X, p.ln1, T0, 512);
      {
        long tot = NR * 512;
        conv_swish_kernel<<<(unsigned)((tot + 255) / 256), 256, 0, stream>>>(
            T0, 512, p.conv_w, p.conv_b, T2, 512, 512, 512, tot);
      }
      const float* gws[4] = {p.i_w, p.f_w, p.z_w, p.o_w};
      for (int g4 = 0; g4 < 4; g4++) {
        const float* src = (g4 < 2) ? T2 : T0;
        for (int h = 0; h < 4; h++)
          gemm(stream, src + h * 128, 512, gws[g4] + (long)h * 128 * 128, 128,
               T1 + g4 * 512 + h * 128, 2048, (int)NR, 128, 128, 0);
      }
      slstm_scan_kernel<<<32, 128, 0, stream>>>(T1, p.R, p.cell_b, Hb, 512);
      slstm_norm_add_kernel<<<(unsigned)(NR * 4), 32, 0, stream>>>(Hb, p.gn_w, X);
      // FFN
      norm_kernel<<<(unsigned)NR, 32, 0, stream>>>(X, p.ln2, T0, 512);
      gemm(stream, T0, 512, p.up, 512, T1, 2048, (int)NR, 2048, 512, 0);
      {
        long tot = NR * 1024;
        ffn_gate_kernel<<<(unsigned)((tot + 255) / 256), 256, 0, stream>>>(T1, T2, tot);
      }
      gemm(stream, T2, 1024, p.down, 1024, X, 512, (int)NR, 512, 1024, 1);
    }
  }
  norm_kernel<<<(unsigned)NR, 32, 0, stream>>>(X, post_norm, (float*)d_out, 512);
}
